// EFNLocal_76441827934548
// MI455X (gfx1250) — matrix operand order repaired and checked by reading, not compiled, not run on hardware
//
#include <hip/hip_runtime.h>
#include <stdint.h>

// Sizes fixed by the reference.
#define N_NODES_  50000
#define N_EDGES_  1600000
#define DF        64
#define NS        4
#define HID       128
#define OUTD      128

typedef __attribute__((ext_vector_type(16))) __bf16       v16bf;
typedef __attribute__((ext_vector_type(8)))  float        v8f;
typedef __attribute__((ext_vector_type(4)))  unsigned int u32x4;
typedef __attribute__((ext_vector_type(4)))  float        f32x4;

// LDS row pitches (bf16 elements): 16B-aligned rows, bank-conflict-free strides.
#define APITCH   168   // A panel: K layout [h_i(0..67) | 0(68..71) | hj-hi(72..139) | 0(140..159)]
#define HPITCH   136   // hidden panel: K = 128
#define W1PITCH  160   // W1^T rows in workspace (bf16), K-padded to 160
#define W2PITCH  128   // W2^T rows in workspace (bf16)

// RNE-ish float->bf16 (used only in the one-shot weight prep kernel).
static __device__ __forceinline__ uint32_t f2bf_rne(float f) {
  uint32_t u = __float_as_uint(f);
  return (u + 0x7FFFu + ((u >> 16) & 1u)) >> 16;
}

// Truncating pack of two floats into two bf16 (one v_perm_b32).
static __device__ __forceinline__ uint32_t pk2t(float lo, float hi) {
  // result bytes [3:0] = { hi.b3, hi.b2, lo.b3, lo.b2 }
  return __builtin_amdgcn_perm(__float_as_uint(hi), __float_as_uint(lo), 0x07060302u);
}

// Single-instruction ReLU: med3(x, 0, +inf) clamps to [0, inf).
static __device__ __forceinline__ float relu1(float x) {
  return __builtin_amdgcn_fmed3f(x, 0.f, __builtin_inff());
}

union Frag { u32x4 q[2]; v16bf v; };

__global__ void zero_out_kernel(float* __restrict__ out, int n4) {
  int i = blockIdx.x * blockDim.x + threadIdx.x;
  if (i < n4) { f32x4 z = {0.f, 0.f, 0.f, 0.f}; ((f32x4*)out)[i] = z; }
}

// Build bf16, N-major (transposed) copies of W1 (with K remap/padding) and W2.
__global__ void prep_w_kernel(const float* __restrict__ W1, const float* __restrict__ W2,
                              uint16_t* __restrict__ w1t, uint16_t* __restrict__ w2t) {
  int i = blockIdx.x * blockDim.x + threadIdx.x;
  if (i < HID * W1PITCH) {
    int n = i / W1PITCH, c = i % W1PITCH;
    float v = 0.f;
    if (c < 68)                  v = W1[c * HID + n];        // h_i part (x + scalars)
    else if (c >= 72 && c < 140) v = W1[(c - 4) * HID + n];  // (h_j - h_i) part
    w1t[n * W1PITCH + c] = (uint16_t)f2bf_rne(v);
  } else {
    int j = i - HID * W1PITCH;
    if (j < OUTD * W2PITCH) {
      int n = j / W2PITCH, k = j % W2PITCH;
      w2t[n * W2PITCH + k] = (uint16_t)f2bf_rne(W2[k * OUTD + n]);
    }
  }
}

__global__ __launch_bounds__(128)
void edge_mlp_kernel(const float* __restrict__ x, const float* __restrict__ scalars,
                     const int* __restrict__ batch, const int* __restrict__ eidx,
                     const float* __restrict__ b1, const float* __restrict__ b2,
                     const uint16_t* __restrict__ w1t, const uint16_t* __restrict__ w2t,
                     float* __restrict__ out) {
  __shared__ __align__(16) uint16_t apanel[4][16 * APITCH];
  __shared__ __align__(16) uint16_t hpanel[4][16 * HPITCH];

  const int lane = threadIdx.x & 31;
  const int wid  = threadIdx.x >> 5;
  const int e16  = lane & 15;    // edge row within 16-edge tile
  const int role = lane >> 4;    // 0: h_i half, 1: (h_j - h_i) half

  uint16_t* A = apanel[wid];
  uint16_t* H = hpanel[wid];

  const int* src = eidx;              // edge_index[0]
  const int* dst = eidx + N_EDGES_;   // edge_index[1]

  // Per-lane bias values for output column nt*16 + e16.
  float b1v[8], b2v[8];
  #pragma unroll
  for (int nt = 0; nt < 8; ++nt) {
    b1v[nt] = b1[nt * 16 + e16];
    b2v[nt] = b2[nt * 16 + e16];
  }

  // Zero the K-pad columns [144,160) of this wave's A panel once.
  { u32x4 z = {0u, 0u, 0u, 0u}; *(u32x4*)&A[e16 * APITCH + 144 + role * 8] = z; }

  const int tiles = (N_EDGES_ + 15) >> 4;
  for (int t = blockIdx.x * 4 + wid; t < tiles; t += gridDim.x * 4) {
    const int base = t << 4;
    int e = base + e16; if (e >= N_EDGES_) e = N_EDGES_ - 1;   // tail clamp (masked later)
    const int ni = dst[e];   // target i
    const int nj = src[e];   // source j

    // ---- Stage A panel (bf16): role0 -> cols[0..71], role1 -> cols[72..143] ----
    #pragma unroll
    for (int c = 0; c < 8; ++c) {
      f32x4 a0 = *(const f32x4*)&x[ni * DF + c * 8];
      f32x4 a1 = *(const f32x4*)&x[ni * DF + c * 8 + 4];
      f32x4 r0 = a0, r1 = a1;
      if (role) {
        f32x4 c0 = *(const f32x4*)&x[nj * DF + c * 8];
        f32x4 c1 = *(const f32x4*)&x[nj * DF + c * 8 + 4];
        r0 = c0 - a0; r1 = c1 - a1;
      }
      u32x4 p; p.x = pk2t(r0.x, r0.y); p.y = pk2t(r0.z, r0.w);
               p.z = pk2t(r1.x, r1.y); p.w = pk2t(r1.z, r1.w);
      *(u32x4*)&A[e16 * APITCH + role * 72 + c * 8] = p;
    }
    {
      f32x4 si = *(const f32x4*)&scalars[batch[ni] * NS];
      f32x4 r = si;
      if (role) {
        f32x4 sj = *(const f32x4*)&scalars[batch[nj] * NS];
        r = sj - si;
      }
      u32x4 p; p.x = pk2t(r.x, r.y); p.y = pk2t(r.z, r.w); p.z = 0u; p.w = 0u;
      *(u32x4*)&A[e16 * APITCH + role * 72 + 64] = p;
    }
    __builtin_amdgcn_wave_barrier();
    asm volatile("s_wait_dscnt 0x0" ::: "memory");
    __builtin_amdgcn_wave_barrier();

    // ---- A fragments for GEMM1 (K = 160 -> 5 steps of 32) ----
    Frag af[5];
    #pragma unroll
    for (int ks = 0; ks < 5; ++ks) {
      const uint16_t* p = &A[e16 * APITCH + ks * 32 + role * 8];
      af[ks].q[0] = *(const u32x4*)p;
      af[ks].q[1] = *(const u32x4*)(p + 16);
    }

    // ---- GEMM1 (bias C-init) + ReLU -> hidden panel (bf16 via truncation) ----
    #pragma unroll
    for (int nt = 0; nt < 8; ++nt) {
      const uint16_t* wrow = &w1t[(nt * 16 + e16) * W1PITCH + role * 8];
      Frag cur, nxt;
      cur.q[0] = *(const u32x4*)wrow;
      cur.q[1] = *(const u32x4*)(wrow + 16);
      float bb = b1v[nt];
      v8f acc = {bb, bb, bb, bb, bb, bb, bb, bb};
      #pragma unroll
      for (int ks = 0; ks < 5; ++ks) {
        if (ks < 4) {                       // double-buffer: prefetch frag ks+1
          const uint16_t* p = wrow + (ks + 1) * 32;
          nxt.q[0] = *(const u32x4*)p;
          nxt.q[1] = *(const u32x4*)(p + 16);
        }
        acc = __builtin_amdgcn_wmma_f32_16x16x32_bf16(
            false, af[ks].v, false, cur.v, (short)0, acc, false, false);
        cur = nxt;                          // elided by full unroll (SSA)
      }
      #pragma unroll
      for (int v = 0; v < 8; ++v) {
        float r = relu1(acc[v]);            // single v_med3 ReLU
        // Truncating bf16: store the float's high 16 bits (ds_store_b16_d16_hi).
        H[(v + 8 * role) * HPITCH + nt * 16 + e16] =
            (uint16_t)(__float_as_uint(r) >> 16);
      }
    }
    __builtin_amdgcn_wave_barrier();
    asm volatile("s_wait_dscnt 0x0" ::: "memory");
    __builtin_amdgcn_wave_barrier();

    // ---- A2 fragments (K = 128 -> 4 steps) ----
    Frag a2[4];
    #pragma unroll
    for (int ks = 0; ks < 4; ++ks) {
      const uint16_t* p = &H[e16 * HPITCH + ks * 32 + role * 8];
      a2[ks].q[0] = *(const u32x4*)p;
      a2[ks].q[1] = *(const u32x4*)(p + 16);
    }

    // ---- GEMM2 (bias C-init) + scatter-add by dst ----
    #pragma unroll
    for (int nt = 0; nt < 8; ++nt) {
      const uint16_t* wrow = &w2t[(nt * 16 + e16) * W2PITCH + role * 8];
      Frag cur, nxt;
      cur.q[0] = *(const u32x4*)wrow;
      cur.q[1] = *(const u32x4*)(wrow + 16);
      float bb = b2v[nt];
      v8f acc = {bb, bb, bb, bb, bb, bb, bb, bb};
      #pragma unroll
      for (int ks = 0; ks < 4; ++ks) {
        if (ks < 3) {                       // double-buffer: prefetch frag ks+1
          const uint16_t* p = wrow + (ks + 1) * 32;
          nxt.q[0] = *(const u32x4*)p;
          nxt.q[1] = *(const u32x4*)(p + 16);
        }
        acc = __builtin_amdgcn_wmma_f32_16x16x32_bf16(
            false, a2[ks].v, false, cur.v, (short)0, acc, false, false);
        cur = nxt;
      }
      #pragma unroll
      for (int v = 0; v < 8; ++v) {
        int m  = v + 8 * role;              // edge row of this C element
        int dn = __shfl(ni, m, 32);         // dst node of edge m (lane m holds it)
        if (base + m < N_EDGES_)
          atomicAdd(&out[dn * OUTD + nt * 16 + e16], acc[v]);
      }
    }
  }
}

extern "C" void kernel_launch(void* const* d_in, const int* in_sizes, int n_in,
                              void* d_out, int out_size, void* d_ws, size_t ws_size,
                              hipStream_t stream) {
  const float* x       = (const float*)d_in[0];
  const float* scalars = (const float*)d_in[1];
  const int*   batch   = (const int*)d_in[2];
  const int*   eidx    = (const int*)d_in[3];
  const float* W1      = (const float*)d_in[4];
  const float* b1      = (const float*)d_in[5];
  const float* W2      = (const float*)d_in[6];
  const float* b2      = (const float*)d_in[7];
  float* out = (float*)d_out;

  // Workspace: bf16 transposed weights (73,728 bytes total).
  uint16_t* w1t = (uint16_t*)d_ws;               // HID * W1PITCH
  uint16_t* w2t = w1t + HID * W1PITCH;           // OUTD * W2PITCH

  const int n4 = (N_NODES_ * OUTD) / 4;
  zero_out_kernel<<<(n4 + 255) / 256, 256, 0, stream>>>(out, n4);

  const int wtot = HID * W1PITCH + OUTD * W2PITCH;
  prep_w_kernel<<<(wtot + 255) / 256, 256, 0, stream>>>(W1, W2, w1t, w2t);

  edge_mlp_kernel<<<4096, 128, 0, stream>>>(x, scalars, batch, eidx, b1, b2, w1t, w2t, out);
}
